// AttentionFilter_69346541961291
// MI455X (gfx1250) — matrix operand order repaired and checked by reading, not compiled, not run on hardware
//
#include <hip/hip_runtime.h>

// ---------------------------------------------------------------------------
// B=64, N=512, D=1024, H=2048
// out = [ filter_matrix (B*N*N floats) | attn_weights (B*N floats) ]
//
// v5 = v4 with the hot-loop W1 fp32->bf16x2 conversion done by a single
// v_perm_b32 per pair (truncation) -- cuts the per-chunk VALU stream ~5x so
// instruction issue stays under the XDL busy time.
// Tiling: 4 waves/block, each wave owns 32 rows x 16 cols (two row-groups
// sharing one B fragment -> 48B/lane LDS per WMMA, 4 independent WMMA chains).
// x tile (128 rows, bf16) + one 16-row W1 chunk live in LDS (~292KB < 320KB).
// ---------------------------------------------------------------------------
#define Bsz   64
#define Nn    512
#define Dd    1024
#define Hh    2048
#define MTILE 128                      // 4 waves x 32 rows
#define NTHR  128
#define NHC   (Hh / 16)                // 128 h-chunks

#define ROWPITCH   (Dd + 8)            // ushorts; 2064B = 129*16B: odd multiple
                                       // of 16B => conflict-free b128 access
#define XS_BYTES   (MTILE * ROWPITCH * 2)            // 264192 B
#define W1S_BYTES  (16 * ROWPITCH * 2)               // 33024 B
#define OFF_XS     0
#define OFF_W1S    (OFF_XS + XS_BYTES)
#define OFF_SCS    (OFF_W1S + W1S_BYTES)
#define OFF_ATT    (OFF_SCS + MTILE * 4)
#define SMEM_BYTES (OFF_ATT + MTILE * 4)             // 298240 B < 320KB

typedef float  v8f   __attribute__((ext_vector_type(8)));
typedef __bf16 v16bf __attribute__((ext_vector_type(16)));

union Frag { v16bf v; uint4 q[2]; };

// round-to-nearest-even pack (used once for the x tile)
__device__ __forceinline__ unsigned int pk_bf16_rne(float a, float b) {
  unsigned int ua = __float_as_uint(a);
  ua += 0x7FFFu + ((ua >> 16) & 1u);
  unsigned int ub = __float_as_uint(b);
  ub += 0x7FFFu + ((ub >> 16) & 1u);
  return (ua >> 16) | (ub & 0xFFFF0000u);
}
// single v_perm_b32 truncating pack (hot W1 path):
// dst[15:0] = a[31:16], dst[31:16] = b[31:16]
__device__ __forceinline__ unsigned int pk_bf16_fast(float a, float b) {
  return __builtin_amdgcn_perm(__float_as_uint(b), __float_as_uint(a),
                               0x07060302u);
}
__device__ __forceinline__ float sigmoidf_(float x) {
  return 1.0f / (1.0f + __expf(-x));
}

extern "C" __global__ __launch_bounds__(NTHR, 1)
void attnfilter_wmma_bf16_v5(const float* __restrict__ x,
                             const float* __restrict__ W1,
                             const float* __restrict__ b1,
                             const float* __restrict__ W2,
                             const float* __restrict__ b2,
                             float* __restrict__ out)
{
  extern __shared__ char smem[];
  unsigned short* xs   = (unsigned short*)(smem + OFF_XS);
  unsigned short* w1s  = (unsigned short*)(smem + OFF_W1S);
  float*          scS  = (float*)(smem + OFF_SCS);
  float*          attS = (float*)(smem + OFF_ATT);

  const int tid  = threadIdx.x;
  const int wave = tid >> 5;         // 0..3 -> rows wave*32 .. wave*32+31
  const int lane = tid & 31;
  const int nc   = lane & 15;        // A-row-in-group / B-column index
  const int kg   = lane >> 4;        // K-group (ISA 7.12.2 layouts)
  const int m0   = blockIdx.x * MTILE;

  // ---- stage x tile fp32 -> bf16 (RNE) into LDS (128 thr x 8 floats/row)
  {
    const int c = tid * 8;
    #pragma unroll 8
    for (int r = 0; r < MTILE; ++r) {
      float4 f0 = *(const float4*)(x + (size_t)(m0 + r) * Dd + c);
      float4 f1 = *(const float4*)(x + (size_t)(m0 + r) * Dd + c + 4);
      uint4 p;
      p.x = pk_bf16_rne(f0.x, f0.y);  p.y = pk_bf16_rne(f0.z, f0.w);
      p.z = pk_bf16_rne(f1.x, f1.y);  p.w = pk_bf16_rne(f1.z, f1.w);
      *(uint4*)(xs + r * ROWPITCH + c) = p;
    }
  }

  // ---- preload W1 chunk 0, converted to packed bf16 (64 VGPRs)
  uint2 w1p[32];
  #pragma unroll
  for (int j = 0; j < 32; ++j) {
    const int r = j >> 1, part = j & 1;
    float4 v = *(const float4*)(W1 + (size_t)r * Dd + part * 512 + tid * 4);
    w1p[j].x = pk_bf16_fast(v.x, v.y);
    w1p[j].y = pk_bf16_fast(v.z, v.w);
  }
  __syncthreads();

  // score partials: element i -> row wave*32 + kg*8 + i (lo) / +16 (hi)
  v8f sAcc0 = {}, sAcc1 = {};

  const unsigned short* aRow0 = xs + (wave * 32 + nc) * ROWPITCH;
  const unsigned short* aRow1 = aRow0 + 16 * ROWPITCH;
  const unsigned short* bRow  = w1s + nc * ROWPITCH;

  for (int hc = 0; hc < NHC; ++hc) {
    // commit staged chunk to LDS
    #pragma unroll
    for (int j = 0; j < 32; ++j) {
      const int r = j >> 1, part = j & 1;
      *(uint2*)(w1s + r * ROWPITCH + part * 512 + tid * 4) = w1p[j];
    }
    __syncthreads();

    // software pipeline: fetch + convert next chunk during the WMMA loop
    if (hc + 1 < NHC) {
      const float* src = W1 + (size_t)(hc + 1) * 16 * Dd + tid * 4;
      #pragma unroll
      for (int j = 0; j < 32; ++j) {
        const int r = j >> 1, part = j & 1;
        float4 v = *(const float4*)(src + (size_t)r * Dd + part * 512);
        w1p[j].x = pk_bf16_fast(v.x, v.y);
        w1p[j].y = pk_bf16_fast(v.z, v.w);
      }
    }

    // K loop: 4 independent WMMA chains; B fragment shared by both row-groups
    v8f a0 = {}, a1 = {}, a2 = {}, a3 = {};
    #pragma unroll
    for (int kp = 0; kp < 8; ++kp) {
      const int k0 = kp * 64;                      // two k-steps: k0, k0+32
      Frag A00, A10, A01, A11, B0, B1;
      A00.q[0] = *(const uint4*)(aRow0 + k0      + kg * 8);
      A00.q[1] = *(const uint4*)(aRow0 + k0 + 16 + kg * 8);
      A10.q[0] = *(const uint4*)(aRow1 + k0      + kg * 8);
      A10.q[1] = *(const uint4*)(aRow1 + k0 + 16 + kg * 8);
      B0.q[0]  = *(const uint4*)(bRow  + k0      + kg * 8);
      B0.q[1]  = *(const uint4*)(bRow  + k0 + 16 + kg * 8);
      A01.q[0] = *(const uint4*)(aRow0 + k0 + 32 + kg * 8);
      A01.q[1] = *(const uint4*)(aRow0 + k0 + 48 + kg * 8);
      A11.q[0] = *(const uint4*)(aRow1 + k0 + 32 + kg * 8);
      A11.q[1] = *(const uint4*)(aRow1 + k0 + 48 + kg * 8);
      B1.q[0]  = *(const uint4*)(bRow  + k0 + 32 + kg * 8);
      B1.q[1]  = *(const uint4*)(bRow  + k0 + 48 + kg * 8);
      a0 = __builtin_amdgcn_wmma_f32_16x16x32_bf16(
               false, A00.v, false, B0.v, (short)0, a0, false, false);
      a1 = __builtin_amdgcn_wmma_f32_16x16x32_bf16(
               false, A10.v, false, B0.v, (short)0, a1, false, false);
      a2 = __builtin_amdgcn_wmma_f32_16x16x32_bf16(
               false, A01.v, false, B1.v, (short)0, a2, false, false);
      a3 = __builtin_amdgcn_wmma_f32_16x16x32_bf16(
               false, A11.v, false, B1.v, (short)0, a3, false, false);
    }

    // fused GEMM2 epilogue: relu(h + b1) * W2, reduced over H in registers
    const int h = hc * 16 + nc;
    const float bias = b1[h];
    const float w2v  = W2[h];
    #pragma unroll
    for (int i = 0; i < 8; ++i) {
      float lo = a0[i] + a2[i] + bias;
      float hi = a1[i] + a3[i] + bias;
      lo = fmaxf(lo, 0.0f);
      hi = fmaxf(hi, 0.0f);
      sAcc0[i] += lo * w2v;
      sAcc1[i] += hi * w2v;
    }
    __syncthreads();   // protect w1s before next chunk's commit
  }

  // ---- reduce score over the 16 column-lanes (stays inside 16-halves)
  #pragma unroll
  for (int mask = 1; mask <= 8; mask <<= 1) {
    #pragma unroll
    for (int i = 0; i < 8; ++i) {
      sAcc0[i] += __shfl_xor((float)sAcc0[i], mask, 32);
      sAcc1[i] += __shfl_xor((float)sAcc1[i], mask, 32);
    }
  }
  if (nc == 0) {
    #pragma unroll
    for (int i = 0; i < 8; ++i) {
      scS[wave * 32 + kg * 8 + i]      = sAcc0[i];
      scS[wave * 32 + 16 + kg * 8 + i] = sAcc1[i];
    }
  }

  // ---- energy gate: one row per thread from the bf16 x tile in LDS
  float e = 0.0f;
  {
    const unsigned short* er = xs + tid * ROWPITCH;
    #pragma unroll 8
    for (int j = 0; j < Dd; j += 8) {
      uint4 q = *(const uint4*)(er + j);
      unsigned int u[4] = {q.x, q.y, q.z, q.w};
      #pragma unroll
      for (int t = 0; t < 4; ++t) {
        float lo = __uint_as_float(u[t] << 16);
        float hi = __uint_as_float(u[t] & 0xFFFF0000u);
        e += lo * lo + hi * hi;
      }
    }
  }
  __syncthreads();

  // ---- combine gates, emit attn_weights (row = tid)
  {
    const float attn = sigmoidf_(scS[tid] + b2[0]) * sigmoidf_(e);
    attS[tid] = attn;
    out[(size_t)Bsz * Nn * Nn + (m0 + tid)] = attn;
  }
  __syncthreads();

  // ---- write filter rows: zeros + diagonal element (coalesced float4)
  #pragma unroll 4
  for (int it = 0; it < (MTILE * Nn) / (NTHR * 4); ++it) {  // 128 iters
    const int idx = it * NTHR + tid;
    const int r   = idx >> 7;            // 128 float4s per row
    const int c4  = (idx & 127) << 2;
    const int m   = m0 + r;
    const int nd  = m & (Nn - 1);        // diag column = m % N
    float4 vv = {0.0f, 0.0f, 0.0f, 0.0f};
    if (nd >= c4 && nd < c4 + 4)
      (&vv.x)[nd - c4] = attS[r];
    *(float4*)(out + ((size_t)m * Nn + c4)) = vv;
  }
}

extern "C" void kernel_launch(void* const* d_in, const int* in_sizes, int n_in,
                              void* d_out, int out_size, void* d_ws, size_t ws_size,
                              hipStream_t stream) {
  (void)in_sizes; (void)n_in; (void)out_size; (void)d_ws; (void)ws_size;
  const float* x  = (const float*)d_in[0];
  const float* W1 = (const float*)d_in[1];
  const float* b1 = (const float*)d_in[2];
  const float* W2 = (const float*)d_in[3];
  const float* b2 = (const float*)d_in[4];
  float* out = (float*)d_out;

  dim3 grid((Bsz * Nn) / MTILE);   // 256 blocks, each owns an M-tile + all H
  dim3 block(NTHR);                // 4 wave32s
  attnfilter_wmma_bf16_v5<<<grid, block, SMEM_BYTES, stream>>>(x, W1, b1, W2, b2, out);
}
